// LSTMModel_30691836297312
// MI455X (gfx1250) — matrix operand order repaired and checked by reading, not compiled, not run on hardware
//
#include <hip/hip_runtime.h>

#define HID   32
#define T_LEN 1024

typedef __attribute__((ext_vector_type(16))) _Float16 v16h;
typedef __attribute__((ext_vector_type(8)))  float    v8f;
typedef __attribute__((ext_vector_type(2)))  float    v2f;

__device__ __forceinline__ float tanh_fast(float x) {
    return __builtin_amdgcn_tanhf(x);           // native v_tanh_f32 (gfx1250)
}
// sigmoid(2a) = 0.5 + 0.5*tanh(a); caller supplies a = 0.5*preactivation
// (the 0.5 is folded into the WMMA constant operands for i/f/o gate tiles)
__device__ __forceinline__ float sigmoid_of_half(float a) {
    return fmaf(0.5f, __builtin_amdgcn_tanhf(a), 0.5f);
}

// One wave handles a 16-batch tile for all T steps.
// Formulation: gates^T[128 x 16] = W_hh[128x32] @ h^T[32x16]  (+ rank-1 x-proj via K=4 f32 WMMA)
//   A tiles  : W_hh rows, f16 A-layout, 8 tiles of 16x32  (constant; i/f/o tiles pre-scaled x0.5)
//   B        : h^T, f16 B-layout (lane n = L%16 = batch, halves = K = 16*(L/16)+h)
//   C/D      : lane = batch col, (VGPR r, L/16) = gate row -> elementwise state aligned
__global__ __launch_bounds__(64)
void lstm_wmma_kernel(const float* __restrict__ x,
                      const float* __restrict__ W_ih,
                      const float* __restrict__ W_hh,
                      const float* __restrict__ b_ih,
                      const float* __restrict__ b_hh,
                      const float* __restrict__ fc_W,
                      const float* __restrict__ fc_b,
                      float* __restrict__ out,
                      int Bsz)
{
    const int lane = threadIdx.x & 31;
    const int wave = (blockIdx.x * blockDim.x + threadIdx.x) >> 5;
    const int b0   = wave * 16;       // batch tile base
    const int n    = lane & 15;       // batch within tile (N column)
    const int hl   = lane >> 4;       // lane half

    if (b0 >= Bsz) return;            // wave-uniform: EXEC stays all-ones for WMMA

    // ---- W_hh as 8 f16 A-layout tiles (16 gates x K=32 each) ----
    // A 16x32 f16, this code's contiguous k order (B matches): lanes 0-15 (m=L): halves 0..15 = K0..15
    //                     lanes 16-31     : halves 0..15 = K16..31
    // Gate tiles: i=0,1  f=2,3  g=4,5  o=6,7. Sigmoid tiles pre-scaled by 0.5 so the
    // WMMA directly produces 0.5*preact, feeding sigmoid(x)=0.5+0.5*tanh(0.5x) with no mul.
    v16h Wa[8];
    #pragma unroll
    for (int gt = 0; gt < 8; ++gt) {
        const float scale = (gt == 4 || gt == 5) ? 1.0f : 0.5f;   // g-gate unscaled
        const int gate = gt * 16 + n;
        const float* wr = W_hh + gate * HID;
        const int k0 = hl * 16;
        #pragma unroll
        for (int i = 0; i < 8; ++i) {
            Wa[gt][i]     = (_Float16)(scale * wr[k0 + i]);
            Wa[gt][8 + i] = (_Float16)(scale * wr[k0 + 8 + i]);
        }
    }

    // ---- rank-1 input projection + bias as f32 K=4 A tiles: A = [w_ih, bias, 0, 0] ----
    // f32 A 16x4: VGPR0 = K0 (lanes<16) / K2 (lanes>=16); VGPR1 = K1 / K3
    v2f Aib[8];
    #pragma unroll
    for (int gt = 0; gt < 8; ++gt) {
        const float scale = (gt == 4 || gt == 5) ? 1.0f : 0.5f;
        const int gate = gt * 16 + n;
        const float w  = scale * W_ih[gate];
        const float bb = scale * (b_ih[gate] + b_hh[gate]);
        Aib[gt][0] = hl ? 0.0f : w;
        Aib[gt][1] = hl ? 0.0f : bb;
    }

    // ---- recurrent state in registers ----
    v16h Bh;                       // h^T in f16 B-layout
    #pragma unroll
    for (int i = 0; i < 16; ++i) Bh[i] = (_Float16)0.0f;
    v8f c[2];                      // cell state, (hidden j, batch) C-layout, j = 16*jt + r + 8*hl
    #pragma unroll
    for (int r = 0; r < 8; ++r) { c[0][r] = 0.0f; c[1][r] = 0.0f; }
    float h0[8], h1[8];            // f32 h of the two j-tiles (for final head)
    #pragma unroll
    for (int r = 0; r < 8; ++r) { h0[r] = 0.0f; h1[r] = 0.0f; }

    const float* xrow = x + (size_t)(b0 + n) * T_LEN;   // D==1: x[b][t]

    for (int tc = 0; tc < T_LEN; tc += 8) {
        const float4 xa = *(const float4*)(xrow + tc);
        const float4 xb = *(const float4*)(xrow + tc + 4);
        __builtin_prefetch(xrow + tc + 64, 0, 1);       // speculative prefetch, 8 chunks ahead
        const float xs[8] = {xa.x, xa.y, xa.z, xa.w, xb.x, xb.y, xb.z, xb.w};

        #pragma unroll
        for (int u = 0; u < 8; ++u) {
            // B for the rank-1 proj: row K0 = x[b], row K1 = 1 (bias), K2/K3 = 0
            v2f Bx;
            Bx[0] = hl ? 1.0f : xs[u];
            Bx[1] = 0.0f;

            v8f zero;
            #pragma unroll
            for (int r = 0; r < 8; ++r) zero[r] = 0.0f;

            // gates^T = W_ih*x + bias  (f32 K=4 WMMA)  then  += W_hh @ h^T  (f16 K=32 WMMA)
            v8f acc[8];
            #pragma unroll
            for (int gt = 0; gt < 8; ++gt) {
                v8f t0 = __builtin_amdgcn_wmma_f32_16x16x4_f32(
                    false, Aib[gt], false, Bx, (short)0, zero, false, false);
                acc[gt] = __builtin_amdgcn_wmma_f32_16x16x32_f16(
                    false, Wa[gt], false, Bh, (short)0, t0, false, false);
            }

            // elementwise LSTM cell; sigmoid tiles already carry the 0.5 pre-scale
            #pragma unroll
            for (int r = 0; r < 8; ++r) {
                {   // jt = 0
                    const float ig = sigmoid_of_half(acc[0][r]);
                    const float fg = sigmoid_of_half(acc[2][r]);
                    const float gg = tanh_fast      (acc[4][r]);
                    const float og = sigmoid_of_half(acc[6][r]);
                    const float cc = fg * c[0][r] + ig * gg;
                    c[0][r] = cc;
                    h0[r]   = og * tanh_fast(cc);
                }
                {   // jt = 1
                    const float ig = sigmoid_of_half(acc[1][r]);
                    const float fg = sigmoid_of_half(acc[3][r]);
                    const float gg = tanh_fast      (acc[5][r]);
                    const float og = sigmoid_of_half(acc[7][r]);
                    const float cc = fg * c[1][r] + ig * gg;
                    c[1][r] = cc;
                    h1[r]   = og * tanh_fast(cc);
                }
            }

            // C-layout f32 h  ->  B-layout f16 h^T : pure lane-pair (L ^ 16) exchange
            //  half r    (K = 16*hl + r)    : hl==0 -> own h0[r] ; hl==1 -> h1[r] from lane L-16
            //  half r+8  (K = 16*hl + 8 + r): hl==0 -> h0[r] from lane L+16 ; hl==1 -> own h1[r]
            #pragma unroll
            for (int r = 0; r < 8; ++r) {
                const float o0 = __shfl_xor(h0[r], 16, 32);
                const float o1 = __shfl_xor(h1[r], 16, 32);
                const float lo = hl ? o1    : h0[r];
                const float hi = hl ? h1[r] : o0;
                Bh[r]     = (_Float16)lo;
                Bh[8 + r] = (_Float16)hi;
            }
        }
    }

    // ---- head: out[b] = h_last[b] . fc_W + fc_b ; lane holds j = 16*jt + r + 8*hl ----
    float partial = 0.0f;
    #pragma unroll
    for (int r = 0; r < 8; ++r) {
        partial += h0[r] * fc_W[r + 8 * hl];
        partial += h1[r] * fc_W[16 + r + 8 * hl];
    }
    partial += __shfl_xor(partial, 16, 32);   // combine the two j-halves (same batch column)
    if (hl == 0 && (b0 + n) < Bsz) {
        out[b0 + n] = partial + fc_b[0];
    }
}

extern "C" void kernel_launch(void* const* d_in, const int* in_sizes, int n_in,
                              void* d_out, int out_size, void* d_ws, size_t ws_size,
                              hipStream_t stream) {
    const float* x    = (const float*)d_in[0];
    const float* W_ih = (const float*)d_in[1];
    const float* W_hh = (const float*)d_in[2];
    const float* b_ih = (const float*)d_in[3];
    const float* b_hh = (const float*)d_in[4];
    const float* fc_W = (const float*)d_in[5];
    const float* fc_b = (const float*)d_in[6];
    float* out = (float*)d_out;

    const int Bsz   = in_sizes[0] / T_LEN;     // D == 1
    const int waves = (Bsz + 15) / 16;         // one wave per 16-batch tile
    const int threads = 64;                    // 2 waves / block -> spread over many WGPs
    const int blocks  = (waves * 32 + threads - 1) / threads;

    lstm_wmma_kernel<<<blocks, threads, 0, stream>>>(
        x, W_ih, W_hh, b_ih, b_hh, fc_W, fc_b, out, Bsz);
}
